// EarthSpecificLayer_9191230014127
// MI455X (gfx1250) — matrix operand order repaired and checked by reading, not compiled, not run on hardware
//
#include <hip/hip_runtime.h>
#include <hip/hip_bf16.h>
#include <math.h>
#include <stdint.h>

// ---------------------------------------------------------------------------
// Pangu EarthSpecificLayer for MI455X (gfx1250), wave32 + WMMA f16->f32.
// Geometry constants (match reference):
//   Z=8 H=96 W=180, window (2,6,12) -> WP=144, WN=960, DIM=384, HEADS=12,
//   DH=32, SHIFT=(1,3,6), NBIAS=3312, WT=64, tokens L=138240.
// ---------------------------------------------------------------------------

typedef _Float16 f16;
typedef __attribute__((ext_vector_type(16))) _Float16 v16h;
typedef __attribute__((ext_vector_type(8)))  float    v8f;
typedef unsigned long long ull;

#define L_TOK   138240
#define DIMC    384
#define WPC     144
#define WNC     960
#define HEADSC  12
#define DHC     32

__device__ __forceinline__ v8f wmma16(v16h a, v16h b, v8f c) {
  return __builtin_amdgcn_wmma_f32_16x16x32_f16(false, a, false, b, (short)0, c,
                                                false, false);
}

// ---- CDNA5 async global->LDS DMA (ASYNCcnt-tracked), 16B per lane ----------
// LDS offset = low 32 bits of the generic pointer (LDS aperture truncation,
// ISA 10.2).  GV addressing mode: saddr = off.
__device__ __forceinline__ void async_copy_b128(const void* gsrc, void* ldst) {
  unsigned loff = (unsigned)(uintptr_t)ldst;
  unsigned long long ga = (unsigned long long)(uintptr_t)gsrc;
  asm volatile("global_load_async_to_lds_b128 %0, %1, off"
               :: "v"(loff), "v"(ga) : "memory");
}
__device__ __forceinline__ void wait_async0() {
  asm volatile("s_wait_asynccnt 0x0" ::: "memory");
}

// A-fragment (16x32 f16): lane l -> row l%16, halves [kb..kb+8) and [16+kb..),
// kb = (l/16)*8.  `stride` in halves; all offsets stay 16B-aligned.
__device__ __forceinline__ v16h ld_fragA(const f16* p, int stride) {
  int lane = threadIdx.x & 31;
  const f16* q = p + (lane & 15) * stride + ((lane >> 4) << 3);
  union { uint4 u[2]; v16h v; } r;
  r.u[0] = *(const uint4*)(q);
  r.u[1] = *(const uint4*)(q + 16);
  return r.v;
}

// B-fragment (32x16 f16): lane l -> column l%16, halves [kb..kb+8) and [16+kb..), kb = (l/16)*8.
__device__ __forceinline__ v16h ld_fragB(const f16* p, int stride) {
  int lane = threadIdx.x & 31;
  const f16* q = p + (lane & 15) * stride + ((lane >> 4) << 3);
  union { uint4 u[2]; v16h v; } r;
  r.u[0] = *(const uint4*)(q);
  r.u[1] = *(const uint4*)(q + 16);
  return r.v;
}

__device__ __forceinline__ float red16_add(float v) {
  v += __shfl_xor(v, 1, 32);
  v += __shfl_xor(v, 2, 32);
  v += __shfl_xor(v, 4, 32);
  v += __shfl_xor(v, 8, 32);
  return v;
}
__device__ __forceinline__ float red16_max(float v) {
  v = fmaxf(v, __shfl_xor(v, 1, 32));
  v = fmaxf(v, __shfl_xor(v, 2, 32));
  v = fmaxf(v, __shfl_xor(v, 4, 32));
  v = fmaxf(v, __shfl_xor(v, 8, 32));
  return v;
}

// windowed token m -> global token index (applies window partition + roll map)
__device__ __forceinline__ int win_to_global(int m) {
  int win = m / WPC, l = m % WPC;
  int mz = win / 240, wr = win % 240;      // MH*MW = 16*15
  int mh = wr / 15,  mw = wr % 15;
  int lz = l / 72,   lr = l % 72;          // WH*WW = 72
  int lh = lr / 12,  lw = lr % 12;
  int z = (mz * 2 + lz + 1) % 8;           // + SHIFT
  int h = (mh * 6 + lh + 3) % 96;
  int w = (mw * 12 + lw + 6) % 180;
  return (z * 96 + h) * 180 + w;
}

// ---------------------------------------------------------------------------
// Kernel 1: QKV GEMM with windowed gather.  Grid (9 nblk, 2160 mblk), 256 thr.
// Writes Q (scaled), K, V as f16 in (win, head, token, d) layout.
// ---------------------------------------------------------------------------
__global__ __launch_bounds__(256) void k_qkv(
    const float* __restrict__ x, const float* __restrict__ wqkv,
    f16* __restrict__ qo, f16* __restrict__ ko, f16* __restrict__ vo) {
  __shared__ f16 As[64 * 32];
  __shared__ f16 Bs[128 * 32];
  int m0 = blockIdx.y * 64, n0 = blockIdx.x * 128;
  int t = threadIdx.x, wave = t >> 5, lane = t & 31;
  int wm = wave >> 2, wn = wave & 3;

  int arow = t >> 2;
  int akk  = (t & 3) << 3;
  size_t ag = (size_t)win_to_global(m0 + arow) * DIMC;
  int krow = t >> 3, nn = (t & 7) << 4;

  v8f acc[2][2] = {};
  for (int kc = 0; kc < 12; ++kc) {
    {  // stage A (gather + f32->f16)
      const float* src = x + ag + kc * 32 + akk;
      float4 p0 = *(const float4*)(src);
      float4 p1 = *(const float4*)(src + 4);
      f16* dst = As + arow * 32 + akk;
      dst[0] = (f16)p0.x; dst[1] = (f16)p0.y; dst[2] = (f16)p0.z; dst[3] = (f16)p0.w;
      dst[4] = (f16)p1.x; dst[5] = (f16)p1.y; dst[6] = (f16)p1.z; dst[7] = (f16)p1.w;
    }
    {  // stage B transposed: Bs[n][k]
      const float* src = wqkv + (size_t)(kc * 32 + krow) * 1152 + n0 + nn;
      if (kc < 11) __builtin_prefetch(src + 32 * 1152, 0, 3);
#pragma unroll
      for (int i = 0; i < 16; i += 4) {
        float4 p = *(const float4*)(src + i);
        Bs[(nn + i + 0) * 32 + krow] = (f16)p.x;
        Bs[(nn + i + 1) * 32 + krow] = (f16)p.y;
        Bs[(nn + i + 2) * 32 + krow] = (f16)p.z;
        Bs[(nn + i + 3) * 32 + krow] = (f16)p.w;
      }
    }
    __syncthreads();
    v16h a0 = ld_fragA(As + (wm * 32 + 0)  * 32, 32);
    v16h a1 = ld_fragA(As + (wm * 32 + 16) * 32, 32);
    v16h b0 = ld_fragB(Bs + (wn * 32 + 0)  * 32, 32);
    v16h b1 = ld_fragB(Bs + (wn * 32 + 16) * 32, 32);
    acc[0][0] = wmma16(a0, b0, acc[0][0]);
    acc[0][1] = wmma16(a0, b1, acc[0][1]);
    acc[1][0] = wmma16(a1, b0, acc[1][0]);
    acc[1][1] = wmma16(a1, b1, acc[1][1]);
    __syncthreads();
  }
  const float qscale = 0.17677669529663687f;  // DH^-0.5
#pragma unroll
  for (int tm = 0; tm < 2; ++tm)
#pragma unroll
    for (int tn = 0; tn < 2; ++tn) {
      int col  = n0 + wn * 32 + tn * 16 + (lane & 15);
      int type = col / DIMC;
      int hd   = (col % DIMC) >> 5;
      int d    = col & 31;
      f16* outp = (type == 0) ? qo : ((type == 1) ? ko : vo);
      float sc  = (type == 0) ? qscale : 1.0f;
#pragma unroll
      for (int r = 0; r < 8; ++r) {
        int m = m0 + wm * 32 + tm * 16 + r + ((lane >> 4) << 3);
        int win = m / WPC, ll = m % WPC;
        outp[(((size_t)win * HEADSC + hd) * WPC + ll) * DHC + d] =
            (f16)(acc[tm][tn][r] * sc);
      }
    }
}

// ---------------------------------------------------------------------------
// Kernel 2: attention per (window, head).  96 threads = 3 waves; each wave
// owns complete 16-row strips -> wave-local softmax.  K of P@V padded to 160.
// Q/K staged via CDNA5 async global->LDS DMA.
// ---------------------------------------------------------------------------
__global__ __launch_bounds__(96) void k_attn(
    const f16* __restrict__ q, const f16* __restrict__ k,
    const f16* __restrict__ v, const float* __restrict__ bias_table,
    f16* __restrict__ attn_out) {
  __shared__ f16 Qs[WPC * 32];
  __shared__ f16 Ks[WPC * 32];
  __shared__ f16 Vt[32 * 160];       // V^T, K padded 144->160 with zeros
  __shared__ f16 Ps[3 * 16 * 160];   // per-wave P strip
  __shared__ int lab[WPC];

  int win = blockIdx.x, head = blockIdx.y;
  int t = threadIdx.x;
  int mz = win / 240, wr = win % 240, mh = wr / 15, mw = wr % 15;
  int wt = mz * 16 + mh;
  size_t base = ((size_t)win * HEADSC + head) * (WPC * DHC);

  {  // stage Q, K via async DMA (9216 B each = 576 x 16B chunks)
    const char* qg = (const char*)(q + base);
    const char* kg = (const char*)(k + base);
#pragma unroll
    for (int j = 0; j < 6; ++j) {
      int i = t + j * 96;
      async_copy_b128(qg + i * 16, (char*)Qs + i * 16);
      async_copy_b128(kg + i * 16, (char*)Ks + i * 16);
    }
    // V transposed (+ zero pad), P pad, shifted-window region labels
    for (int i = t; i < WPC * 32; i += 96) {
      int row = i >> 5, d = i & 31;
      Vt[d * 160 + row] = v[base + i];
    }
    for (int i = t; i < 32 * 16; i += 96)
      Vt[(i >> 4) * 160 + 144 + (i & 15)] = (f16)0.f;
    for (int i = t; i < 3 * 16 * 16; i += 96)
      Ps[(i >> 4) * 160 + 144 + (i & 15)] = (f16)0.f;
    for (int i = t; i < WPC; i += 96) {
      int lz = i / 72, lh = (i % 72) / 12, lw = i % 12;
      int sz = mz * 2 + lz, sh = mh * 6 + lh, sw = mw * 12 + lw;
      int rz = (sz < 6) ? 0 : ((sz < 7) ? 1 : 2);
      int rh = (sh < 90) ? 0 : ((sh < 93) ? 1 : 2);
      int rw = (sw < 168) ? 0 : ((sw < 174) ? 1 : 2);
      lab[i] = rz * 9 + rh * 3 + rw;
    }
  }
  wait_async0();
  __syncthreads();

  int wave = t >> 5, lane = t & 31;
  int coll = lane & 15, rbase = (lane >> 4) << 3;
  f16* strip = Ps + wave * 16 * 160;

  for (int mt = wave; mt < 9; mt += 3) {          // 3 uniform iterations/wave
    v16h aQ = ld_fragA(Qs + mt * 16 * 32, 32);
    v8f s[9];
#pragma unroll
    for (int jt = 0; jt < 9; ++jt) {
      v16h bK = ld_fragB(Ks + jt * 16 * 32, 32);
      v8f zz = {};
      s[jt] = wmma16(aQ, bK, zz);
    }
    // earth bias + shifted-window mask
#pragma unroll
    for (int r = 0; r < 8; ++r) {
      int i  = mt * 16 + r + rbase;
      int z1 = i / 72, h1 = (i % 72) / 12, w1 = i % 12;
      int li = lab[i];
#pragma unroll
      for (int jt = 0; jt < 9; ++jt) {
        int j  = jt * 16 + coll;
        int z2 = j / 72, h2 = (j % 72) / 12, w2 = j % 12;
        int idx = (z1 + 2 * z2) * 828 + (h1 + 6 * h2) * 23 + (w1 - w2 + 11);
        float b = bias_table[idx * 768 + wt * 12 + head];
        if (lab[j] != li) b -= 100.f;
        s[jt][r] += b;
      }
    }
    // softmax over j (wave-local; rows split across 16-lane half-groups)
#pragma unroll
    for (int r = 0; r < 8; ++r) {
      float mx = -1e30f;
#pragma unroll
      for (int jt = 0; jt < 9; ++jt) mx = fmaxf(mx, s[jt][r]);
      mx = red16_max(mx);
      float sum = 0.f;
#pragma unroll
      for (int jt = 0; jt < 9; ++jt) {
        float e = __expf(s[jt][r] - mx);
        s[jt][r] = e;
        sum += e;
      }
      sum = red16_add(sum);
      float inv = 1.f / sum;
#pragma unroll
      for (int jt = 0; jt < 9; ++jt)
        strip[(r + rbase) * 160 + jt * 16 + coll] = (f16)(s[jt][r] * inv);
    }
    // out = P @ V  (K = 160, zero-padded)
#pragma unroll
    for (int nt = 0; nt < 2; ++nt) {
      v8f o = {};
#pragma unroll
      for (int kc = 0; kc < 5; ++kc) {
        v16h aP = ld_fragA(strip + kc * 32, 160);
        v16h bV = ld_fragB(Vt + (nt * 16) * 160 + kc * 32, 160);
        o = wmma16(aP, bV, o);
      }
      int d = nt * 16 + coll;
#pragma unroll
      for (int r = 0; r < 8; ++r) {
        int ll = mt * 16 + r + rbase;
        attn_out[(((size_t)win * WPC + ll) * HEADSC + head) * DHC + d] = (f16)o[r];
      }
    }
  }
}

// ---------------------------------------------------------------------------
// Kernel 3: proj GEMM (384x384) + window-reverse scatter + LN1 + residual.
// Grid 2160, 256 threads; A tiles staged via async global->LDS DMA.
// ---------------------------------------------------------------------------
__global__ __launch_bounds__(256) void k_proj(
    const f16* __restrict__ attn, const float* __restrict__ wproj,
    const float* __restrict__ bproj, const float* __restrict__ x,
    const float* __restrict__ ln1w, const float* __restrict__ ln1b,
    float* __restrict__ x1) {
  __shared__ f16 As[64 * 32];
  __shared__ f16 Bs[384 * 32];
  __shared__ float rsum[64], rsq[64];
  int m0 = blockIdx.x * 64;
  int t = threadIdx.x, wave = t >> 5, lane = t & 31;
  int wm = wave >> 2, wn = wave & 3;
  int coll = lane & 15, rb = (lane >> 4) << 3;
  int krow = t >> 3, nn0 = (t & 7) << 4;

  v8f acc[2][6] = {};
  for (int kc = 0; kc < 12; ++kc) {
    {  // A: pure f16 copy -> async DMA to LDS
      int row = t >> 2, kk = (t & 3) << 3;
      async_copy_b128(attn + (size_t)(m0 + row) * DIMC + kc * 32 + kk,
                      As + row * 32 + kk);
    }
#pragma unroll
    for (int nb = 0; nb < 3; ++nb) {  // B transposed: Bs[n][k]
      int nn = nb * 128 + nn0;
      const float* src = wproj + (size_t)(kc * 32 + krow) * DIMC + nn;
      if (kc < 11) __builtin_prefetch(src + 32 * DIMC, 0, 3);
#pragma unroll
      for (int i = 0; i < 16; i += 4) {
        float4 p = *(const float4*)(src + i);
        Bs[(nn + i + 0) * 32 + krow] = (f16)p.x;
        Bs[(nn + i + 1) * 32 + krow] = (f16)p.y;
        Bs[(nn + i + 2) * 32 + krow] = (f16)p.z;
        Bs[(nn + i + 3) * 32 + krow] = (f16)p.w;
      }
    }
    wait_async0();
    __syncthreads();
    v16h a0 = ld_fragA(As + (wm * 32 + 0)  * 32, 32);
    v16h a1 = ld_fragA(As + (wm * 32 + 16) * 32, 32);
#pragma unroll
    for (int tn = 0; tn < 6; ++tn) {
      v16h b = ld_fragB(Bs + (wn * 96 + tn * 16) * 32, 32);
      acc[0][tn] = wmma16(a0, b, acc[0][tn]);
      acc[1][tn] = wmma16(a1, b, acc[1][tn]);
    }
    __syncthreads();
  }
  if (t < 64) { rsum[t] = 0.f; rsq[t] = 0.f; }
  __syncthreads();
#pragma unroll
  for (int tm = 0; tm < 2; ++tm)
#pragma unroll
    for (int tn = 0; tn < 6; ++tn) {
      float bp = bproj[wn * 96 + tn * 16 + coll];
#pragma unroll
      for (int r = 0; r < 8; ++r) acc[tm][tn][r] += bp;
    }
#pragma unroll
  for (int tm = 0; tm < 2; ++tm)
#pragma unroll
    for (int r = 0; r < 8; ++r) {
      float s = 0.f, q2 = 0.f;
#pragma unroll
      for (int tn = 0; tn < 6; ++tn) {
        float vv = acc[tm][tn][r];
        s += vv; q2 += vv * vv;
      }
      s = red16_add(s); q2 = red16_add(q2);
      if (coll == 0) {
        int row = wm * 32 + tm * 16 + r + rb;
        atomicAdd(&rsum[row], s);
        atomicAdd(&rsq[row], q2);
      }
    }
  __syncthreads();
#pragma unroll
  for (int tm = 0; tm < 2; ++tm)
#pragma unroll
    for (int r = 0; r < 8; ++r) {
      int row = wm * 32 + tm * 16 + r + rb;
      size_t g = (size_t)win_to_global(m0 + row);
      float mean = rsum[row] * (1.f / 384.f);
      float var  = rsq[row]  * (1.f / 384.f) - mean * mean;
      float inv  = rsqrtf(var + 1e-5f);
#pragma unroll
      for (int tn = 0; tn < 6; ++tn) {
        int n = wn * 96 + tn * 16 + coll;
        float vln = (acc[tm][tn][r] - mean) * inv * ln1w[n] + ln1b[n];
        x1[g * DIMC + n] = x[g * DIMC + n] + vln;
      }
    }
}

// ---------------------------------------------------------------------------
// Kernel 4: fused MLP (384->1536->384, exact GELU) + LN2 + residual.
// y2 (64x384) accumulates in registers across 12 hidden chunks of 128.
// ---------------------------------------------------------------------------
__global__ __launch_bounds__(256) void k_mlp(
    const float* __restrict__ x1, const float* __restrict__ w1,
    const float* __restrict__ b1, const float* __restrict__ w2,
    const float* __restrict__ b2, const float* __restrict__ ln2w,
    const float* __restrict__ ln2b, float* __restrict__ out) {
  __shared__ f16 As[64 * 32];
  __shared__ f16 B1s[128 * 32];
  __shared__ f16 Hs[64 * 128];
  __shared__ f16 B2s[384 * 32];
  __shared__ float rsum[64], rsq[64];
  int m0 = blockIdx.x * 64;
  int t = threadIdx.x, wave = t >> 5, lane = t & 31;
  int wm = wave >> 2, wn = wave & 3;
  int coll = lane & 15, rb = (lane >> 4) << 3;
  int krow = t >> 3, nn0 = (t & 7) << 4;

  v8f y[2][6] = {};
  for (int hc = 0; hc < 12; ++hc) {
    v8f g1[2][2] = {};
    for (int kc = 0; kc < 12; ++kc) {
      {  // A from x1 (f32->f16)
        int row = t >> 2, kk = (t & 3) << 3;
        const float* src = x1 + (size_t)(m0 + row) * DIMC + kc * 32 + kk;
        float4 p0 = *(const float4*)(src);
        float4 p1 = *(const float4*)(src + 4);
        f16* dst = As + row * 32 + kk;
        dst[0] = (f16)p0.x; dst[1] = (f16)p0.y; dst[2] = (f16)p0.z; dst[3] = (f16)p0.w;
        dst[4] = (f16)p1.x; dst[5] = (f16)p1.y; dst[6] = (f16)p1.z; dst[7] = (f16)p1.w;
      }
      {  // B1 transposed: B1s[n][k]
        const float* src = w1 + (size_t)(kc * 32 + krow) * 1536 + hc * 128 + nn0;
        if (kc < 11) __builtin_prefetch(src + 32 * 1536, 0, 3);
#pragma unroll
        for (int i = 0; i < 16; i += 4) {
          float4 p = *(const float4*)(src + i);
          B1s[(nn0 + i + 0) * 32 + krow] = (f16)p.x;
          B1s[(nn0 + i + 1) * 32 + krow] = (f16)p.y;
          B1s[(nn0 + i + 2) * 32 + krow] = (f16)p.z;
          B1s[(nn0 + i + 3) * 32 + krow] = (f16)p.w;
        }
      }
      __syncthreads();
      v16h a0 = ld_fragA(As + (wm * 32 + 0)  * 32, 32);
      v16h a1 = ld_fragA(As + (wm * 32 + 16) * 32, 32);
      v16h b0 = ld_fragB(B1s + (wn * 32 + 0)  * 32, 32);
      v16h b1v = ld_fragB(B1s + (wn * 32 + 16) * 32, 32);
      g1[0][0] = wmma16(a0, b0,  g1[0][0]);
      g1[0][1] = wmma16(a0, b1v, g1[0][1]);
      g1[1][0] = wmma16(a1, b0,  g1[1][0]);
      g1[1][1] = wmma16(a1, b1v, g1[1][1]);
      __syncthreads();
    }
    // bias + exact GELU -> Hs (f16)
#pragma unroll
    for (int tm = 0; tm < 2; ++tm)
#pragma unroll
      for (int tn = 0; tn < 2; ++tn) {
        int nl = wn * 32 + tn * 16 + coll;
        float bb = b1[hc * 128 + nl];
#pragma unroll
        for (int r = 0; r < 8; ++r) {
          float vv = g1[tm][tn][r] + bb;
          vv = 0.5f * vv * (1.f + erff(vv * 0.7071067811865476f));
          Hs[(wm * 32 + tm * 16 + r + rb) * 128 + nl] = (f16)vv;
        }
      }
    __syncthreads();
    // GEMM2 partial: y += Hs(64x128) @ w2[hc*128 .. , :]
    for (int kc = 0; kc < 4; ++kc) {
#pragma unroll
      for (int nb = 0; nb < 3; ++nb) {  // B2 transposed: B2s[n][k]
        int nn = nb * 128 + nn0;
        const float* src = w2 + (size_t)(hc * 128 + kc * 32 + krow) * DIMC + nn;
        if (kc < 3) __builtin_prefetch(src + 32 * DIMC, 0, 3);
#pragma unroll
        for (int i = 0; i < 16; i += 4) {
          float4 p = *(const float4*)(src + i);
          B2s[(nn + i + 0) * 32 + krow] = (f16)p.x;
          B2s[(nn + i + 1) * 32 + krow] = (f16)p.y;
          B2s[(nn + i + 2) * 32 + krow] = (f16)p.z;
          B2s[(nn + i + 3) * 32 + krow] = (f16)p.w;
        }
      }
      __syncthreads();
      v16h a0 = ld_fragA(Hs + (wm * 32 + 0)  * 128 + kc * 32, 128);
      v16h a1 = ld_fragA(Hs + (wm * 32 + 16) * 128 + kc * 32, 128);
#pragma unroll
      for (int tn = 0; tn < 6; ++tn) {
        v16h b = ld_fragB(B2s + (wn * 96 + tn * 16) * 32, 32);
        y[0][tn] = wmma16(a0, b, y[0][tn]);
        y[1][tn] = wmma16(a1, b, y[1][tn]);
      }
      __syncthreads();
    }
  }
  // + b2, LN2, residual -> out
  if (t < 64) { rsum[t] = 0.f; rsq[t] = 0.f; }
  __syncthreads();
#pragma unroll
  for (int tm = 0; tm < 2; ++tm)
#pragma unroll
    for (int tn = 0; tn < 6; ++tn) {
      float bp = b2[wn * 96 + tn * 16 + coll];
#pragma unroll
      for (int r = 0; r < 8; ++r) y[tm][tn][r] += bp;
    }
#pragma unroll
  for (int tm = 0; tm < 2; ++tm)
#pragma unroll
    for (int r = 0; r < 8; ++r) {
      float s = 0.f, q2 = 0.f;
#pragma unroll
      for (int tn = 0; tn < 6; ++tn) {
        float vv = y[tm][tn][r];
        s += vv; q2 += vv * vv;
      }
      s = red16_add(s); q2 = red16_add(q2);
      if (coll == 0) {
        int row = wm * 32 + tm * 16 + r + rb;
        atomicAdd(&rsum[row], s);
        atomicAdd(&rsq[row], q2);
      }
    }
  __syncthreads();
#pragma unroll
  for (int tm = 0; tm < 2; ++tm)
#pragma unroll
    for (int r = 0; r < 8; ++r) {
      int row = wm * 32 + tm * 16 + r + rb;
      size_t g = (size_t)(m0 + row);
      float mean = rsum[row] * (1.f / 384.f);
      float var  = rsq[row]  * (1.f / 384.f) - mean * mean;
      float inv  = rsqrtf(var + 1e-5f);
#pragma unroll
      for (int tn = 0; tn < 6; ++tn) {
        int n = wn * 96 + tn * 16 + coll;
        float vln = (y[tm][tn][r] - mean) * inv * ln2w[n] + ln2b[n];
        out[g * DIMC + n] = x1[g * DIMC + n] + vln;
      }
    }
}

// ---------------------------------------------------------------------------
extern "C" void kernel_launch(void* const* d_in, const int* in_sizes, int n_in,
                              void* d_out, int out_size, void* d_ws,
                              size_t ws_size, hipStream_t stream) {
  (void)in_sizes; (void)n_in; (void)out_size; (void)ws_size;
  const float* x          = (const float*)d_in[0];
  const float* wqkv       = (const float*)d_in[1];
  const float* wproj      = (const float*)d_in[2];
  const float* bproj      = (const float*)d_in[3];
  const float* bias_table = (const float*)d_in[4];
  const float* ln1w       = (const float*)d_in[5];
  const float* ln1b       = (const float*)d_in[6];
  const float* ln2w       = (const float*)d_in[7];
  const float* ln2b       = (const float*)d_in[8];
  const float* w1         = (const float*)d_in[9];
  const float* b1         = (const float*)d_in[10];
  const float* w2         = (const float*)d_in[11];
  const float* b2         = (const float*)d_in[12];
  float* out = (float*)d_out;

  // workspace: 4 f16 activation buffers + 1 f32 (total ~636 MB)
  const size_t E = (size_t)L_TOK * DIMC;
  char* ws = (char*)d_ws;
  f16* qws = (f16*)ws;  ws += E * sizeof(f16);
  f16* kws = (f16*)ws;  ws += E * sizeof(f16);
  f16* vws = (f16*)ws;  ws += E * sizeof(f16);
  f16* aws = (f16*)ws;  ws += E * sizeof(f16);
  float* x1 = (float*)ws;

  dim3 g1(9, 2160);
  k_qkv<<<g1, 256, 0, stream>>>(x, wqkv, qws, kws, vws);
  dim3 g2(WNC, HEADSC);
  k_attn<<<g2, 96, 0, stream>>>(qws, kws, vws, bias_table, aws);
  k_proj<<<2160, 256, 0, stream>>>(aws, wproj, bproj, x, ln1w, ln1b, x1);
  k_mlp<<<2160, 256, 0, stream>>>(x1, w1, b1, w2, b2, ln2w, ln2b, out);
}
